// QuanvolutionEnhanced_65481071405616
// MI455X (gfx1250) — matrix operand order repaired and checked by reading, not compiled, not run on hardware
//
#include <hip/hip_runtime.h>
#include <hip/hip_bf16.h>

typedef __attribute__((ext_vector_type(16))) _Float16 v16h;
typedef __attribute__((ext_vector_type(8)))  _Float16 v8h;
typedef __attribute__((ext_vector_type(8)))  float    v8f;

#define THREADS        256
#define WAVES          8
#define ROWS_PER_BLOCK 128          // 8 waves x 16 rows
#define KCHUNKS        25           // 25 * 32 = 800 >= 784
#define ROWSTRIDE      808          // halves; 1616B = 404 dwords -> conflict-free b128 reads
#define TILE_HALVES    (16 * ROWSTRIDE)
#define WLANE_STRIDE   24           // halves (48B) per lane -> 2-way (minimum) on b128
#define WCHUNK_STRIDE  (32 * WLANE_STRIDE)

__global__ __launch_bounds__(THREADS)
void quanv_head_kernel(const float* __restrict__ x,
                       const float* __restrict__ W,
                       const float* __restrict__ bias,
                       float* __restrict__ out)
{
    __shared__ _Float16 sFeat[WAVES * TILE_HALVES];      // 206848 B
    __shared__ _Float16 sW[KCHUNKS * WCHUNK_STRIDE];     //  38400 B

    const int tid      = threadIdx.x;
    const int rowBlock = blockIdx.x * ROWS_PER_BLOCK;

    // ---- Phase 1a: stage W into B-fragment layout (f16, classes 10..15 zero) ----
    for (int idx = tid; idx < KCHUNKS * 32; idx += THREADS) {
        const int kc  = idx >> 5;
        const int l   = idx & 31;
        const int n   = l & 15;           // class column
        const int khi = l >> 4;           // 0: K {0..7,16..23}, 1: K {8..15,24..31} of chunk
        const int kb  = kc * 32 + khi * 8;
        _Float16* dst = &sW[kc * WCHUNK_STRIDE + l * WLANE_STRIDE];
        #pragma unroll
        for (int j = 0; j < 16; ++j) {
            const int K = kb + (j < 8 ? j : j + 8);
            const float v = (n < 10 && K < 784) ? W[n * 784 + K] : 0.0f;
            dst[j] = (_Float16)v;
        }
    }

    // ---- Phase 1b: zero the K padding (784..807) for every row ----
    for (int idx = tid; idx < ROWS_PER_BLOCK * (ROWSTRIDE - 784); idx += THREADS) {
        const int r = idx / (ROWSTRIDE - 784);
        const int k = 784 + idx % (ROWSTRIDE - 784);
        sFeat[(r >> 4) * TILE_HALVES + (r & 15) * ROWSTRIDE + k] = (_Float16)0.0f;
    }

    // ---- Phase 1c: quanvolution features, coalesced float2 pixel pairs ----
    // pair (x0,x1) at image (i, 2jp..2jp+1) -> feats k0 = 4*(14*(i/2)+jp) + 2*(i&1):
    //   f[k0]   = cos(x0) + x0
    //   f[k0+1] = cos(x0)*cos(x1) + x1
    const int PAIRS = ROWS_PER_BLOCK * 392;
    for (int i = tid; i < PAIRS; i += THREADS) {
        const int rl   = i / 392;       // local row 0..127
        const int pi   = i % 392;       // pair index within row
        const int iimg = pi / 14;       // image row 0..27
        const int jp   = pi % 14;       // pair column 0..13
        const float2 xv = *(const float2*)(x + (size_t)(rowBlock + rl) * 784 + 2 * pi);
        const float c0 = __cosf(xv.x);
        const float f0 = c0 + xv.x;
        const float f1 = c0 * __cosf(xv.y) + xv.y;
        const int p  = 14 * (iimg >> 1) + jp;
        const int k0 = 4 * p + 2 * (iimg & 1);           // even -> 4B-aligned pair store
        _Float16* dst = &sFeat[(rl >> 4) * TILE_HALVES + (rl & 15) * ROWSTRIDE + k0];
        dst[0] = (_Float16)f0;
        dst[1] = (_Float16)f1;
    }

    __syncthreads();

    // ---- Phase 2: per-wave 16x16 logits over K=800 via v_wmma_f32_16x16x32_f16 ----
    const int wave = tid >> 5;
    const int lane = tid & 31;
    const int hi16 = lane >> 4;         // A: 0 -> K {0..7,16..23}, 1 -> K {8..15,24..31}
    const int rl   = lane & 15;         // A row within tile / B column (class)
    const _Float16* ft = &sFeat[wave * TILE_HALVES];

    v8f acc = {};
    #pragma unroll 5
    for (int kc = 0; kc < KCHUNKS; ++kc) {
        const int kb = kc * 32;
        // A fragment: two 16B LDS reads per the 16-bit A-matrix layout
        const _Float16* ap = ft + rl * ROWSTRIDE + kb + hi16 * 8;
        const v8h alo = *(const v8h*)(ap);
        const v8h ahi = *(const v8h*)(ap + 16);
        const v16h a = __builtin_shufflevector(alo, ahi,
            0, 1, 2, 3, 4, 5, 6, 7, 8, 9, 10, 11, 12, 13, 14, 15);
        // B fragment: 32B contiguous per lane
        const _Float16* bp = &sW[kc * WCHUNK_STRIDE + lane * WLANE_STRIDE];
        const v8h blo = *(const v8h*)(bp);
        const v8h bhi = *(const v8h*)(bp + 8);
        const v16h bm = __builtin_shufflevector(blo, bhi,
            0, 1, 2, 3, 4, 5, 6, 7, 8, 9, 10, 11, 12, 13, 14, 15);
        acc = __builtin_amdgcn_wmma_f32_16x16x32_f16(
            false, a, false, bm, (short)0, acc, false, false);
    }

    // ---- Epilogue: bias + log_softmax in registers (butterfly within 16-lane halves) ----
    // C layout: VGPR r, lanes 0-15 -> row r, N=lane; lanes 16-31 -> row r+8, N=lane-16.
    const int   n     = lane & 15;
    const bool  valid = (n < 10);
    const float bv    = valid ? bias[n] : 0.0f;
    const int   rowBase = rowBlock + wave * 16 + hi16 * 8;
    #pragma unroll
    for (int r = 0; r < 8; ++r) {
        const float logit = acc[r] + bv;
        float m = valid ? logit : -__builtin_inff();
        #pragma unroll
        for (int off = 8; off >= 1; off >>= 1)
            m = fmaxf(m, __shfl_xor(m, off, 32));
        float s = valid ? __expf(logit - m) : 0.0f;
        #pragma unroll
        for (int off = 8; off >= 1; off >>= 1)
            s += __shfl_xor(s, off, 32);
        if (valid)
            out[(size_t)(rowBase + r) * 10 + n] = (logit - m) - __logf(s);
    }
}

extern "C" void kernel_launch(void* const* d_in, const int* in_sizes, int n_in,
                              void* d_out, int out_size, void* d_ws, size_t ws_size,
                              hipStream_t stream) {
    (void)n_in; (void)out_size; (void)d_ws; (void)ws_size;
    const float* x = (const float*)d_in[0];
    const float* W = (const float*)d_in[1];
    const float* b = (const float*)d_in[2];
    float* out     = (float*)d_out;

    const int rows   = in_sizes[0] / 784;          // 65536
    const int blocks = rows / ROWS_PER_BLOCK;      // 512
    quanv_head_kernel<<<blocks, THREADS, 0, stream>>>(x, W, b, out);
}
